// ProductKey_36000415875512
// MI455X (gfx1250) — matrix operand order repaired and checked by reading, not compiled, not run on hardware
//
#include <hip/hip_runtime.h>
#include <hip/hip_bf16.h>

// Problem constants (fixed by setup_inputs).
#define B_    4
#define C_    2048
#define H_    8
#define D_    512
#define SUB   256     // D/2, K-dim of each GEMM
#define N_    1024    // subkeys
#define TOPK  32
#define ROWS  32      // query rows per workgroup (two 16-row WMMA M-blocks)
#define NTHREADS 256  // 8 waves (wave32)

typedef __attribute__((ext_vector_type(16))) __bf16        v16bf;
typedef __attribute__((ext_vector_type(8)))  float         v8f;
typedef __attribute__((ext_vector_type(4)))  unsigned int  v4u;

union Frag {
  v16bf        bf;
  unsigned int u[8];
  v4u          q[2];
};

__device__ __forceinline__ unsigned short f32_to_bf16(float f) {
  unsigned int u = __float_as_uint(f);
  u += 0x7FFFu + ((u >> 16) & 1u);   // round-to-nearest-even
  return (unsigned short)(u >> 16);
}

__device__ __forceinline__ v8f wmma_bf16(const Frag& a, const Frag& b, v8f c) {
  return __builtin_amdgcn_wmma_f32_16x16x32_bf16(
      false, a.bf, false, b.bf, (short)0, c, false, false);
}

// ---- Kernel 0: convert keys f32 -> bf16 into workspace ([2][H][N][SUB]) ----
__global__ void pk_convert_keys(const float* __restrict__ kl,
                                const float* __restrict__ kr,
                                unsigned short* __restrict__ out, int n) {
  int i = blockIdx.x * blockDim.x + threadIdx.x;
  const int stride = gridDim.x * blockDim.x;
  for (; i < n; i += stride) {
    out[i]     = f32_to_bf16(kl[i]);
    out[i + n] = f32_to_bf16(kr[i]);
  }
}

// ---- Kernel 1: fused bf16-WMMA scores + two-stage top-k + softmax ----
__global__ __launch_bounds__(NTHREADS, 1) void pk_main(
    const float* __restrict__ q,
    const unsigned short* __restrict__ kbf,   // [2][H][N][SUB] bf16
    float* __restrict__ outw,
    int*   __restrict__ outi)
{
  __shared__ unsigned short qt[ROWS][SUB];            // 16 KB (current side's Q, bf16)
  __shared__ float          sc[ROWS][N_];             // 128 KB score slab (reused per side)
  __shared__ float          topV[2][ROWS][TOPK];      // 8 KB
  __shared__ unsigned short topI[2][ROWS][TOPK];      // 4 KB (indices < 1024)

  const int tile = blockIdx.x;     // 0..255  (M-tile over B*C rows)
  const int h    = blockIdx.y;     // head
  const int tid  = threadIdx.x;
  const int wv   = tid >> 5;       // wave 0..7
  const int ln   = tid & 31;       // lane

  for (int side = 0; side < 2; ++side) {
    // ---------- Phase A: stage this side's query tile in LDS as bf16 ----------
    for (int e = tid; e < ROWS * SUB; e += NTHREADS) {
      const int m = e >> 8;
      const int k = e & (SUB - 1);
      const float v =
          q[((size_t)(tile * ROWS + m) * H_ + h) * D_ + side * SUB + k];
      qt[m][k] = f32_to_bf16(v);
    }
    __syncthreads();   // qt ready; previous top-k done with sc

    // ---------- Phase B: sc = Q_bf16 * K_bf16^T (f32 accum WMMA) ----------
    {
      const unsigned short* kside = kbf + (size_t)side * (H_ * N_ * SUB);
      const int lncol = ln & 15;
      const int koff  = (ln >> 4);       // B layout: hi lanes hold K+8..15, K+24..31
      const int hi8   = (ln >> 4) * 8;   // A layout: hi lanes hold shifted K set
      const int mrow  = (ln >> 4) << 3;  // C/D layout row base

      // Load one full B tile (16 consecutive subkeys x 256 K) into registers.
      auto loadB = [&](Frag* dst, int nt) {
        const v4u* krow =
            (const v4u*)(kside + ((size_t)h * N_ + nt * 16 + lncol) * SUB);
#pragma unroll
        for (int kc = 0; kc < 8; ++kc) {
          dst[kc].q[0] = krow[kc * 4 + koff];
          dst[kc].q[1] = krow[kc * 4 + koff + 2];
        }
      };

      // 16 WMMAs per N-tile: two 16-row M-blocks reuse every B fragment.
      auto computeT = [&](Frag* b, int nt) {
        const int ncol = nt * 16 + lncol;
        const unsigned int* qr0 = (const unsigned int*)&qt[lncol][0];
        const unsigned int* qr1 = (const unsigned int*)&qt[16 + lncol][0];
        v8f acc0 = {}, acc1 = {};
#pragma unroll
        for (int kc = 0; kc < 8; ++kc) {
          const int k0d = (kc * 32 + hi8) >> 1;   // dword index into qt row
          Frag a0, a1;
#pragma unroll
          for (int v = 0; v < 4; ++v) {           // 2 contiguous b128 LDS loads each
            a0.u[v]     = qr0[k0d + v];
            a0.u[4 + v] = qr0[k0d + 8 + v];
            a1.u[v]     = qr1[k0d + v];
            a1.u[4 + v] = qr1[k0d + 8 + v];
          }
          acc0 = wmma_bf16(a0, b[kc], acc0);
          acc1 = wmma_bf16(a1, b[kc], acc1);
        }
#pragma unroll
        for (int p = 0; p < 8; ++p) {
          sc[mrow + p][ncol]      = acc0[p];
          sc[16 + mrow + p][ncol] = acc1[p];
        }
      };

      // Software pipeline: double-buffer B tiles across this wave's 8 N-tiles.
      Frag bA[8], bB[8];
      const int nt0 = wv * 8;
      loadB(bA, nt0);
#pragma unroll 1
      for (int p = 0; p < 4; ++p) {
        loadB(bB, nt0 + 2 * p + 1);
        computeT(bA, nt0 + 2 * p);
        if (p < 3) loadB(bA, nt0 + 2 * p + 2);
        computeT(bB, nt0 + 2 * p + 1);
      }
    }
    __syncthreads();   // score slab complete

    // ---------- Phase C: exact top-32 of 1024 per row (this side) ----------
    for (int t4 = 0; t4 < 4; ++t4) {
      const int m = wv * 4 + t4;         // 32 rows over 8 waves
      float* s = &sc[m][0];

      float cv = -1e30f; int ci = ln;    // lane-local max over strided 32 elems
      for (int j = 0; j < 32; ++j) {
        const float v = s[ln + 32 * j];
        if (v > cv) { cv = v; ci = ln + 32 * j; }
      }
      for (int t = 0; t < TOPK; ++t) {
        float bv = cv; int bl = ln;      // wave argmax (val, lane)
#pragma unroll
        for (int msk = 16; msk; msk >>= 1) {
          const float ov = __shfl_xor(bv, msk, 32);
          const int   ol = __shfl_xor(bl, msk, 32);
          if (ov > bv || (ov == bv && ol < bl)) { bv = ov; bl = ol; }
        }
        const int widx = __shfl(ci, bl, 32);
        if (ln == 0) {
          topV[side][m][t] = bv;
          topI[side][m][t] = (unsigned short)widx;
        }
        if (ln == bl) {                  // winner invalidates & rescans
          s[ci] = -1e30f;
          cv = -1e30f; ci = ln;
          for (int j = 0; j < 32; ++j) {
            const float v = s[ln + 32 * j];
            if (v > cv) { cv = v; ci = ln + 32 * j; }
          }
        }
      }
    }
    __syncthreads();   // topV/topI[side] published; sc free for next side
  }

  // ---------- Phase D: product top-32 via 32-way sorted merge + softmax ----------
  for (int r4 = 0; r4 < 4; ++r4) {
    const int m  = wv * 4 + r4;          // 32 rows over 8 waves
    const float Rv = topV[1][m][ln];     // lane ln owns right candidate ln
    const int   Ri = (int)topI[1][m][ln];
    int   ptr  = 0;                      // pointer into sorted-descending L list
    float head = topV[0][m][0] + Rv;
    float myV  = 0.0f; int myI = 0; float mx = 0.0f;

    for (int t = 0; t < TOPK; ++t) {
      float bv = head; int bl = ln;
#pragma unroll
      for (int msk = 16; msk; msk >>= 1) {
        const float ov = __shfl_xor(bv, msk, 32);
        const int   ol = __shfl_xor(bl, msk, 32);
        if (ov > bv || (ov == bv && ol < bl)) { bv = ov; bl = ol; }
      }
      const int wptr = __shfl(ptr, bl, 32);             // uniform
      const int wRi  = __shfl(Ri,  bl, 32);             // uniform
      const int flat = (int)topI[0][m][wptr] * N_ + wRi; // broadcast LDS read
      if (t == 0) mx = bv;                               // max (descending order)
      if (ln == t) { myV = bv; myI = flat; }             // lane t keeps t-th pick
      if (ln == bl) {                                    // winner advances
        ++ptr;
        head = (ptr < TOPK) ? (topV[0][m][ptr] + Rv) : -1e30f;
      }
    }

    float e = __expf(myV - mx);
    float sum = e;
#pragma unroll
    for (int msk = 16; msk; msk >>= 1) sum += __shfl_xor(sum, msk, 32);

    const size_t o = ((size_t)(tile * ROWS + m) * H_ + h) * TOPK + ln;
    outw[o] = e / sum;
    outi[o] = myI;
  }
}

extern "C" void kernel_launch(void* const* d_in, const int* in_sizes, int n_in,
                              void* d_out, int out_size, void* d_ws, size_t ws_size,
                              hipStream_t stream) {
  (void)in_sizes; (void)n_in; (void)out_size; (void)ws_size;
  const float* q  = (const float*)d_in[0];
  const float* kl = (const float*)d_in[1];
  const float* kr = (const float*)d_in[2];

  unsigned short* kbf = (unsigned short*)d_ws;   // [2][H][N][SUB] bf16 (8 MB)
  const int nk = H_ * N_ * SUB;                  // elems per side
  pk_convert_keys<<<512, 256, 0, stream>>>(kl, kr, kbf, nk);

  float* outw = (float*)d_out;
  int*   outi = (int*)((float*)d_out + (size_t)B_ * C_ * H_ * TOPK);
  dim3 grid((B_ * C_) / ROWS, H_);
  pk_main<<<grid, NTHREADS, 0, stream>>>(q, kbf, outw, outi);
}
